// MultiInterposePufNet_75677323755965
// MI455X (gfx1250) — matrix operand order repaired and checked by reading, not compiled, not run on hardware
//
#include <hip/hip_runtime.h>
#include <math.h>

// ---------------- CDNA5 types ----------------
typedef __attribute__((ext_vector_type(16))) _Float16     v16h;
typedef __attribute__((ext_vector_type(8)))  float        v8f;
typedef __attribute__((ext_vector_type(4)))  unsigned int u32x4;
typedef __attribute__((ext_vector_type(2)))  unsigned int u32x2;

#define B_ROWS   32768
#define N_STAGES 128
#define M_PUF    32
#define NCOL     768      // 256 x-cols | 256 u-cols | 256 v-cols
#define NT       48       // n-tiles of 16
#define KTILES   4        // k-tiles of 32

// workspace layout (bytes):
//   phi16 : B_ROWS*128*2            = 8388608
//   wpack : NT*KTILES*2*256*4       = 393216   (hi/lo f16 fragments, register layout)
//   biasv : NCOL*4                  = 3072
#define WS_PHI   0
#define WS_WPACK 8388608
#define WS_BIAS  (8388608 + 393216)

// ---------------------------------------------------------------------------
// Kernel 1: pack combined weight matrix [768 x 128] into WMMA B-fragment
// register layout, split into f16 hi + f16 lo residual; also build bias vec.
//   rows   0..255 : Wx  (col j of raw_x == row j of Wx)           bias = bx
//   rows 256..511 : WU[mk,k] = Wy[mk,k] (k<=63), Wy[mk,64] (k==64), 0 else
//   rows 512..767 : WV[mk,k] = 0 (k<64), Wy[mk,k+1] (k>=64)       bias = Wy[mk,129]
// B fragment mapping used here (contiguous k, same as A): lane l, dword j: N=l&15,
//   K = ktile*32 + (l<16?0:16) + {2j, 2j+1}   (low half = even K)
// ---------------------------------------------------------------------------
__global__ __launch_bounds__(256) void prep_weights(
    const float* __restrict__ Wx, const float* __restrict__ bx,
    const float* __restrict__ Wy, _Float16* __restrict__ wpack_h,
    float* __restrict__ biasv) {
  int idx = blockIdx.x * blockDim.x + threadIdx.x;
  if (idx >= NCOL * N_STAGES) return;
  int n = idx >> 7;        // 0..767  (output column)
  int k = idx & 127;       // 0..127  (phi index)
  float w;
  if (n < 256) {
    w = Wx[n * 128 + k];
  } else if (n < 512) {
    int mk = n - 256;
    w = (k < 64) ? Wy[mk * 130 + k] : ((k == 64) ? Wy[mk * 130 + 64] : 0.0f);
  } else {
    int mk = n - 512;
    w = (k < 64) ? 0.0f : Wy[mk * 130 + (k + 1)];
  }
  _Float16 hi = (_Float16)w;
  _Float16 lo = (_Float16)(w - (float)hi);
  int t = n >> 4, nl = n & 15;
  int ktile = k >> 5, kr = k & 31;
  int lane = nl + (kr & 16);
  int j = (kr & 15) >> 1;
  int base = ((t * KTILES + ktile) * 2) * 256 + lane * 8 + j; // dword index (hi)
  wpack_h[(base)       * 2 + (kr & 1)] = hi;
  wpack_h[(base + 256) * 2 + (kr & 1)] = lo;
  if (k == 0) {
    biasv[n] = (n < 256) ? bx[n] : ((n < 512) ? 0.0f : Wy[(n - 512) * 130 + 129]);
  }
}

// ---------------------------------------------------------------------------
// Kernel 2: phi[b,i] = prod_{j>=i} (1-2*x[b,j])  — one wave32 per row,
// 4 columns per lane, suffix scan across lanes via shfl. Output f16 (exact ±1).
// ---------------------------------------------------------------------------
__global__ __launch_bounds__(256) void phi_kernel(
    const float* __restrict__ x, _Float16* __restrict__ phi16) {
  int lane = threadIdx.x & 31;
  int row  = blockIdx.x * 8 + (threadIdx.x >> 5);
  float4 x4 = reinterpret_cast<const float4*>(x + (size_t)row * 128)[lane];
  float s0 = 1.0f - 2.0f * x4.x;
  float s1 = 1.0f - 2.0f * x4.y;
  float s2 = 1.0f - 2.0f * x4.z;
  float s3 = 1.0f - 2.0f * x4.w;
  float p = s0 * s1 * s2 * s3;
  // inclusive suffix product across lanes (Hillis-Steele, downward)
  float v = p;
#pragma unroll
  for (int off = 1; off < 32; off <<= 1) {
    float o = __shfl_down(v, off, 32);
    if (lane + off < 32) v *= o;
  }
  float excl = __shfl_down(v, 1, 32);
  if (lane == 31) excl = 1.0f;
  float f3 = s3 * excl;
  float f2 = s2 * f3;
  float f1 = s1 * f2;
  float f0 = s0 * f1;
  union { _Float16 h[4]; u32x2 u; } pk;
  pk.h[0] = (_Float16)f0; pk.h[1] = (_Float16)f1;
  pk.h[2] = (_Float16)f2; pk.h[3] = (_Float16)f3;
  reinterpret_cast<u32x2*>(phi16 + (size_t)row * 128)[lane] = pk.u;
}

// ---------------------------------------------------------------------------
// Kernel 3: fused WMMA GEMM + epilogue.
// Wave owns 16 rows; A = phi rows (4 k-tiles, loaded once, reused for all 48
// n-tiles). Per tile: batch-load ALL B fragments first (loads stay in flight
// across the WMMA chain), then 8/16 chained WMMAs, then a batched cross-lane
// product reduction (8 independent ds_bpermute chains per step -> one wait).
// ---------------------------------------------------------------------------
union AF { v16h h; u32x4 q[2]; };

// product across lanes {l^1, l^2, l^4} for 8 independent values at once
__device__ __forceinline__ void prod8_vec(float (&p)[8]) {
#pragma unroll
  for (int s = 1; s <= 4; s <<= 1) {
    float t[8];
#pragma unroll
    for (int r = 0; r < 8; ++r) t[r] = __shfl_xor(p[r], s, 32);
#pragma unroll
    for (int r = 0; r < 8; ++r) p[r] *= t[r];
  }
}

__global__ __launch_bounds__(256) void fused_puf(
    const _Float16* __restrict__ phi16,
    const unsigned int* __restrict__ wpack,
    const float* __restrict__ biasv,
    float* __restrict__ out) {
  __shared__ float xoLds[8 * 16 * 32];   // per-wave [16 rows][32 pufs]
  const int lane = threadIdx.x & 31;
  const int wv   = threadIdx.x >> 5;
  const int rb   = (blockIdx.x * 8 + wv) * 16;
  const int nl   = lane & 15;
  const int rhalf = (lane < 16) ? 0 : 8;
  const int mgrp  = nl >> 3;             // which 8-col group in tile

  // ---- load A fragments: 16 rows x 128 K of phi, 4 k-tiles ----
  // A 16x32 layout used here (contiguous k, as in wpack): lane l, dword i: M=l&15,
//   K = kt*32 + 2i + (l>=16?16:0)  -> two b128 loads per k-tile
  const u32x4* phiQ = reinterpret_cast<const u32x4*>(phi16);
  const int rowD = (rb + nl) * 64;                 // row base in dwords
  const int khD  = (lane >> 4) << 3;               // +0 or +8 dwords
  AF a[4];
#pragma unroll
  for (int kt = 0; kt < 4; ++kt) {
    int d = rowD + kt * 16 + khD;                  // dword index, 16B aligned
    a[kt].q[0] = phiQ[(d >> 2)];
    a[kt].q[1] = phiQ[(d >> 2) + 1];
  }

  const u32x4* wq = reinterpret_cast<const u32x4*>(wpack);
  const int laneQ = lane * 2;                      // lane offset in u32x4 units

  // -------- phase 1: x-arbiters (n-tiles 0..15) --------
#pragma unroll 1
  for (int t = 0; t < 16; ++t) {
    AF bf[8];
#pragma unroll
    for (int kt = 0; kt < 4; ++kt) {
      int fb = (t * 4 + kt) * 128 + laneQ;         // hi fragment (u32x4 idx)
      bf[2 * kt    ].q[0] = wq[fb];      bf[2 * kt    ].q[1] = wq[fb + 1];
      bf[2 * kt + 1].q[0] = wq[fb + 64]; bf[2 * kt + 1].q[1] = wq[fb + 65];
    }
    float bb = biasv[t * 16 + nl];
    v8f acc = {};
#pragma unroll
    for (int kt = 0; kt < 4; ++kt) {
      acc = __builtin_amdgcn_wmma_f32_16x16x32_f16(false, a[kt].h, false,
                bf[2 * kt].h, (short)0, acc, false, false);
      acc = __builtin_amdgcn_wmma_f32_16x16x32_f16(false, a[kt].h, false,
                bf[2 * kt + 1].h, (short)0, acc, false, false);
    }
    float pv[8];
#pragma unroll
    for (int r = 0; r < 8; ++r) pv[r] = acc[r] + bb;
    prod8_vec(pv);                                 // prod over 8 xor-arbiters
    int mloc = t * 2 + mgrp;
#pragma unroll
    for (int r = 0; r < 8; ++r) {
      float xo = 1.0f - 2.0f / (1.0f + __expf(-pv[r])); // 1 - 2*sigmoid
      if ((lane & 7) == 0) xoLds[wv * 512 + (rhalf + r) * 32 + mloc] = xo;
    }
  }
  __syncthreads();

  // -------- phase 2: y-arbiters (u: tiles 16..31, v: tiles 32..47) --------
#pragma unroll 1
  for (int t = 0; t < 16; ++t) {
    AF bu[8], bv[8];
#pragma unroll
    for (int kt = 0; kt < 4; ++kt) {
      int fbU = ((16 + t) * 4 + kt) * 128 + laneQ;
      int fbV = ((32 + t) * 4 + kt) * 128 + laneQ;
      bu[2 * kt    ].q[0] = wq[fbU];      bu[2 * kt    ].q[1] = wq[fbU + 1];
      bu[2 * kt + 1].q[0] = wq[fbU + 64]; bu[2 * kt + 1].q[1] = wq[fbU + 65];
      bv[2 * kt    ].q[0] = wq[fbV];      bv[2 * kt    ].q[1] = wq[fbV + 1];
      bv[2 * kt + 1].q[0] = wq[fbV + 64]; bv[2 * kt + 1].q[1] = wq[fbV + 65];
    }
    float vb = biasv[512 + t * 16 + nl];
    v8f au = {}, av = {};
#pragma unroll
    for (int kt = 0; kt < 4; ++kt) {
      au = __builtin_amdgcn_wmma_f32_16x16x32_f16(false, a[kt].h, false,
                bu[2 * kt].h, (short)0, au, false, false);
      au = __builtin_amdgcn_wmma_f32_16x16x32_f16(false, a[kt].h, false,
                bu[2 * kt + 1].h, (short)0, au, false, false);
      av = __builtin_amdgcn_wmma_f32_16x16x32_f16(false, a[kt].h, false,
                bv[2 * kt].h, (short)0, av, false, false);
      av = __builtin_amdgcn_wmma_f32_16x16x32_f16(false, a[kt].h, false,
                bv[2 * kt + 1].h, (short)0, av, false, false);
    }
    int mloc = t * 2 + mgrp;
    float val[8];
#pragma unroll
    for (int r = 0; r < 8; ++r) {
      float xo  = xoLds[wv * 512 + (rhalf + r) * 32 + mloc];
      float axo = fabsf(xo);
      val[r] = xo * au[r] + axo * (av[r] + vb);
    }
    prod8_vec(val);                                // prod over 8 y-arbiters
#pragma unroll
    for (int r = 0; r < 8; ++r) {
      if ((lane & 7) == 0)
        out[(size_t)mloc * B_ROWS + rb + rhalf + r] =
            1.0f / (1.0f + __expf(-val[r]));
    }
  }
}

// ---------------------------------------------------------------------------
extern "C" void kernel_launch(void* const* d_in, const int* in_sizes, int n_in,
                              void* d_out, int out_size, void* d_ws, size_t ws_size,
                              hipStream_t stream) {
  (void)in_sizes; (void)n_in; (void)out_size; (void)ws_size;
  const float* x  = (const float*)d_in[0];
  const float* Wx = (const float*)d_in[1];
  const float* bx = (const float*)d_in[2];
  const float* Wy = (const float*)d_in[3];
  float* out = (float*)d_out;

  char* ws = (char*)d_ws;
  _Float16*     phi16 = (_Float16*)(ws + WS_PHI);
  unsigned int* wpack = (unsigned int*)(ws + WS_WPACK);
  float*        biasv = (float*)(ws + WS_BIAS);

  prep_weights<<<(NCOL * N_STAGES + 255) / 256, 256, 0, stream>>>(
      Wx, bx, Wy, (_Float16*)wpack, biasv);
  phi_kernel<<<B_ROWS / 8, 256, 0, stream>>>(x, phi16);
  fused_puf<<<B_ROWS / 128, 256, 0, stream>>>(phi16, wpack, biasv, out);
}
